// ConstractiveLayer_52682068853283
// MI455X (gfx1250) — matrix operand order repaired and checked by reading, not compiled, not run on hardware
//
#include <hip/hip_runtime.h>
#include <stdint.h>

#define N_NODES 50000
#define N_EDGES 800000
#define C 128
#define K_TOP 25000
#define SORT_N 65536
#define WPITCH 136          // LDS pitch in bf16 elems: 272B rows -> 16B aligned, conflict-free
#define TILE_ROWS 64        // GEMM rows per workgroup (4 waves x 16)

typedef __attribute__((ext_vector_type(16))) __bf16 v16bf;
typedef __attribute__((ext_vector_type(8)))  float  v8f;
typedef __attribute__((ext_vector_type(4)))  unsigned su4;
typedef __attribute__((ext_vector_type(8)))  unsigned su8;

static __device__ __forceinline__ __bf16 f2bf(float f) {
    union { float f; unsigned u; } in; in.f = f;
    unsigned u = in.u;
    unsigned r = (u + 0x7FFFu + ((u >> 16) & 1u)) >> 16;  // RNE
    union { unsigned short s; __bf16 b; } out; out.s = (unsigned short)r;
    return out.b;
}

// ---------------------------------------------------------------------------
// TDM descriptor helpers (CDNA5 ISA ch.8: D# group0 = 4 SGPRs, group1 = 8)
// ---------------------------------------------------------------------------
static __device__ __forceinline__ su4 tdm_g0(unsigned lds_addr, unsigned long long gaddr) {
    su4 g;
    g[0] = 1u;                                   // count=1, user mode
    g[1] = lds_addr;                             // LDS byte address
    g[2] = (unsigned)gaddr;                      // global addr [31:0]
    g[3] = ((unsigned)(gaddr >> 32) & 0x01FFFFFFu) | (2u << 30); // addr[56:32] | type=2
    return g;
}

static __device__ __forceinline__ void tdm_issue(su4 g0, su8 g1) {
    asm volatile("tensor_load_to_lds %0, %1" :: "s"(g0), "s"(g1) : "memory");
}

// ---------------------------------------------------------------------------
// GEMM: Out[r,n] = Xbf[r,:] @ W[:,n] (+ bias[n])
//  - Xbf : bf16 row-major [nRows x 128]
//  - Wt  : bf16 pre-transposed + padded global image [128 x WPITCH] (Wt[n][k])
//  Tiles staged into LDS by the Tensor Data Mover; inner loop is ds_load+WMMA.
// ---------------------------------------------------------------------------
__global__ __launch_bounds__(128) void gemm_tdm_wmma(
    const __bf16* __restrict__ Xbf, const __bf16* __restrict__ Wt,
    const float* __restrict__ bias, float* __restrict__ Out, int nRows)
{
    __shared__ __bf16 sW[128 * WPITCH];        // 34816 B
    __shared__ __bf16 sX[TILE_ROWS * WPITCH];  // 17408 B

    const int t = threadIdx.x;
    const int rowBase = blockIdx.x * TILE_ROWS;

    if (t < 32) {   // wave 0 issues both DMA descriptors (EXEC ignored by TDM)
        // --- W: 1-D copy of the padded bf16 weight image (4B units) ---
        {
            const unsigned units = 128 * WPITCH / 2;         // 8704 dwords
            su8 g1;
            g1[0] = (2u << 16);                              // data_size=4B
            g1[1] = (units & 0xFFFFu) << 16;                 // tensor_dim0 lo
            g1[2] = (units >> 16) | (1u << 16);              // tensor_dim0 hi | tensor_dim1=1
            g1[3] = (units << 16);                           // tile_dim0
            g1[4] = 0;                                       // tile_dim1=0 (1-D)
            g1[5] = units;                                   // tensor_dim0_stride
            g1[6] = 0; g1[7] = 0;
            tdm_issue(tdm_g0((unsigned)(uintptr_t)&sW[0],
                             (unsigned long long)(uintptr_t)Wt), g1);
        }
        // --- X tile: 2-D 128 x TILE_ROWS (2B units), LDS padding -> pitch 136,
        //     tensor_dim1 = remaining rows so OOB rows are hardware-zero-filled ---
        {
            const unsigned rowsAvail = (unsigned)(nRows - rowBase);
            su8 g1;
            g1[0] = (1u << 16)                               // data_size=2B
                  | (1u << 20)                               // pad_enable
                  | (5u << 22)                               // pad_interval: 64 dwords (=256B row)
                  | (3u << 25);                              // pad_amount: 4 dwords (=16B = 8 bf16)
            g1[1] = (128u & 0xFFFFu) << 16;                  // tensor_dim0 = 128
            g1[2] = (rowsAvail & 0xFFFFu) << 16;             // tensor_dim1 lo
            g1[3] = (rowsAvail >> 16) | (128u << 16);        // tensor_dim1 hi | tile_dim0=128
            g1[4] = TILE_ROWS;                               // tile_dim1
            g1[5] = 128u;                                    // tensor_dim0_stride
            g1[6] = 0; g1[7] = 0;
            tdm_issue(tdm_g0((unsigned)(uintptr_t)&sX[0],
                             (unsigned long long)(uintptr_t)(Xbf + (size_t)rowBase * C)), g1);
        }
        __builtin_amdgcn_s_wait_tensorcnt(0);
    }
    __syncthreads();

    const int lane = t & 31;
    const int wave = t >> 5;        // 4 waves, 16 rows each
    const int hlf  = lane >> 4;
    const int mn   = lane & 15;

    const __bf16* arow = &sX[(wave * 16 + mn) * WPITCH];
    v8f acc[8] = {};

#pragma unroll
    for (int kb = 0; kb < 4; ++kb) {
        const int k0 = kb * 32;
        // A fragment (16-bit A 16x32): k = k0 + half*16 + e
        v16bf a;
#pragma unroll
        for (int e = 0; e < 8; ++e) a[e]     = arow[k0 + hlf * 16 + e];
#pragma unroll
        for (int e = 0; e < 8; ++e) a[8 + e] = arow[k0 + hlf * 16 + 8 + e];
#pragma unroll
        for (int tc = 0; tc < 8; ++tc) {
            // B fragment (16-bit B 32x16): lane's column, k = k0 + half*16 + e
            const __bf16* bcol = &sW[(tc * 16 + mn) * WPITCH + k0 + hlf * 16];
            v16bf b;
#pragma unroll
            for (int e = 0; e < 16; ++e) b[e] = bcol[e];
            acc[tc] = __builtin_amdgcn_wmma_f32_16x16x32_bf16(
                false, a, false, b, (short)0, acc[tc], false, false);
        }
    }

#pragma unroll
    for (int tc = 0; tc < 8; ++tc) {
        int n = tc * 16 + mn;
        float bn = bias ? bias[n] : 0.0f;
#pragma unroll
        for (int r = 0; r < 8; ++r) {
            int row = rowBase + wave * 16 + hlf * 8 + r;
            if (row < nRows) Out[(size_t)row * C + n] = acc[tc][r] + bn;
        }
    }
}

// ---------------------------------------------------------------------------
// Conversion kernels (bf16 staging for the TDM+WMMA GEMMs)
// ---------------------------------------------------------------------------
__global__ void conv_bf16(const float* __restrict__ in, __bf16* __restrict__ out, int n) {
    int i = blockIdx.x * blockDim.x + threadIdx.x;
    if (i < n) out[i] = f2bf(in[i]);
}

// h = relu(a[c]*y + s[c]) fused into the bf16 conversion
__global__ void conv_affine_bf16(const float* __restrict__ y, const float* __restrict__ a,
                                 const float* __restrict__ s, __bf16* __restrict__ out, int total) {
    int i = blockIdx.x * blockDim.x + threadIdx.x;
    if (i < total) {
        int c = i & (C - 1);
        out[i] = f2bf(fmaxf(0.0f, a[c] * y[i] + s[c]));
    }
}

// Wt[n*WPITCH + k] = bf16(W[k*128 + n]); pad columns zeroed
__global__ void conv_wT(const float* __restrict__ W, __bf16* __restrict__ wt) {
    int i = blockIdx.x * blockDim.x + threadIdx.x;
    if (i < 128 * WPITCH) {
        int n = i / WPITCH, k = i - n * WPITCH;
        wt[i] = (k < 128) ? f2bf(W[k * 128 + n]) : (__bf16)0.0f;
    }
}

// ---------------------------------------------------------------------------
// Utility / reduction kernels
// ---------------------------------------------------------------------------
__global__ void fill_f32(float* p, float v, int n) {
    int i = blockIdx.x * blockDim.x + threadIdx.x;
    if (i < n) p[i] = v;
}
__global__ void fill_i32(int* p, int v, int n) {
    int i = blockIdx.x * blockDim.x + threadIdx.x;
    if (i < n) p[i] = v;
}

__global__ void colstats(const float* __restrict__ M, const float* __restrict__ bias,
                         int nRows, float* __restrict__ sums)
{
    int c = threadIdx.x & 127;
    int r0 = (threadIdx.x >> 7) + blockIdx.x * 2;
    int rs = gridDim.x * 2;
    float b = bias ? bias[c] : 0.0f;
    float s = 0.0f, s2 = 0.0f;
    for (int r = r0; r < nRows; r += rs) {
        float v = M[(size_t)r * C + c] + b;
        s += v; s2 += v * v;
    }
    atomicAdd(&sums[c], s);
    atomicAdd(&sums[C + c], s2);
}

__global__ void bn_params(const float* sums, const float* g, const float* bv,
                          float invN, float* a, float* s)
{
    int c = threadIdx.x;
    float mu  = sums[c] * invN;
    float var = sums[C + c] * invN - mu * mu;
    float ai  = g[c] * rsqrtf(var + 1e-5f);
    a[c] = ai;
    s[c] = bv[c] - mu * ai;
}

__global__ void pool_norm_k(const float* w, float* invnorm) {
    __shared__ float red[C];
    int c = threadIdx.x;
    red[c] = w[c] * w[c];
    __syncthreads();
    for (int o = 64; o > 0; o >>= 1) {
        if (c < o) red[c] += red[c + o];
        __syncthreads();
    }
    if (c == 0) invnorm[0] = rsqrtf(red[0]);
}

// ---------------------------------------------------------------------------
// GCN aggregation
// ---------------------------------------------------------------------------
__global__ void deg_accum(const int* __restrict__ dst, float* deg, int e) {
    int i = blockIdx.x * blockDim.x + threadIdx.x;
    if (i < e) atomicAdd(&deg[dst[i]], 1.0f);
}

__global__ void agg_init(const float* __restrict__ hw, const float* __restrict__ deg,
                         float* __restrict__ agg, int total)
{
    int i = blockIdx.x * blockDim.x + threadIdx.x;
    if (i < total) agg[i] = hw[i] / deg[i >> 7];   // self-loop: dinv^2 = 1/deg
}

__global__ void agg_edges(const int* __restrict__ src, const int* __restrict__ dst,
                          const float* __restrict__ hw, const float* __restrict__ deg,
                          float* __restrict__ agg, int E)
{
    int t = blockIdx.x * blockDim.x + threadIdx.x;
    int e = t >> 5, lane = t & 31;
    if (e >= E) return;
    int r = src[e], c = dst[e];
    float s = rsqrtf(deg[r]) * rsqrtf(deg[c]);
    float4 v = ((const float4*)(hw + (size_t)r * C))[lane];
    float* ap = agg + (size_t)c * C + lane * 4;
    atomicAdd(ap + 0, v.x * s);
    atomicAdd(ap + 1, v.y * s);
    atomicAdd(ap + 2, v.z * s);
    atomicAdd(ap + 3, v.w * s);
}

// h2 = relu(bn2(agg + gcn_b));  score = tanh((h2 . pool_w) * invnorm)
__global__ void h2score(const float* __restrict__ agg, const float* __restrict__ gcn_b,
                        const float* __restrict__ a2, const float* __restrict__ s2,
                        const float* __restrict__ pw, const float* __restrict__ invnorm,
                        float* __restrict__ h2, float* __restrict__ score)
{
    __shared__ float red[C];
    int i = blockIdx.x, c = threadIdx.x;
    float v = agg[(size_t)i * C + c] + gcn_b[c];
    float h = fmaxf(0.0f, a2[c] * v + s2[c]);
    h2[(size_t)i * C + c] = h;
    red[c] = h * pw[c];
    __syncthreads();
    for (int o = 64; o > 0; o >>= 1) {
        if (c < o) red[c] += red[c + o];
        __syncthreads();
    }
    if (c == 0) score[i] = tanhf(red[0] * invnorm[0]);
}

// ---------------------------------------------------------------------------
// Top-K via 64-bit keys + global bitonic sort (descending, jax tie-break)
// ---------------------------------------------------------------------------
__global__ void build_keys(const float* __restrict__ score, unsigned long long* keys) {
    int i = blockIdx.x * blockDim.x + threadIdx.x;
    unsigned long long key = 0ull;
    if (i < N_NODES) {
        union { float f; unsigned u; } s; s.f = score[i];
        unsigned u = s.u;
        unsigned fk = (u & 0x80000000u) ? ~u : (u | 0x80000000u);
        key = ((unsigned long long)fk << 32) |
              (unsigned long long)(0xFFFFFFFFu - (unsigned)i);
    }
    keys[i] = key;
}

__global__ void bitonic_pass(unsigned long long* keys, int j, int k) {
    int i = blockIdx.x * blockDim.x + threadIdx.x;
    int ixj = i ^ j;
    if (ixj > i) {
        unsigned long long a = keys[i], b = keys[ixj];
        bool swap = ((i & k) == 0) ? (a < b) : (a > b);
        if (swap) { keys[i] = b; keys[ixj] = a; }
    }
}

__global__ void select_topk(const unsigned long long* __restrict__ keys,
                            const float* __restrict__ score, const float* __restrict__ h2,
                            const int* __restrict__ batch, float* __restrict__ out,
                            int* __restrict__ new_idx)
{
    int i = blockIdx.x, c = threadIdx.x;
    unsigned idx = 0xFFFFFFFFu - (unsigned)(keys[i] & 0xFFFFFFFFull);
    float ts = score[idx];
    out[(size_t)i * C + c] = h2[(size_t)idx * C + c] * ts;
    if (c == 0) {
        new_idx[idx] = i;
        out[(size_t)K_TOP * C + 2ull * N_EDGES + i] = (float)batch[idx];
    }
}

__global__ void edge_out(const int* __restrict__ src, const int* __restrict__ dst,
                         const int* __restrict__ new_idx, float* __restrict__ out)
{
    int e = blockIdx.x * blockDim.x + threadIdx.x;
    if (e >= N_EDGES) return;
    int a = new_idx[src[e]];
    int b = new_idx[dst[e]];
    bool keep = (a >= 0) && (b >= 0);
    out[(size_t)K_TOP * C + e]           = keep ? (float)a : -1.0f;
    out[(size_t)K_TOP * C + N_EDGES + e] = keep ? (float)b : -1.0f;
}

// ---------------------------------------------------------------------------
extern "C" void kernel_launch(void* const* d_in, const int* in_sizes, int n_in,
                              void* d_out, int out_size, void* d_ws, size_t ws_size,
                              hipStream_t stream)
{
    const float* x      = (const float*)d_in[0];
    const int*   ei     = (const int*)  d_in[1];
    const int*   batch  = (const int*)  d_in[2];
    const float* lin_w  = (const float*)d_in[3];
    const float* lin_b  = (const float*)d_in[4];
    const float* bn1_g  = (const float*)d_in[5];
    const float* bn1_b  = (const float*)d_in[6];
    const float* gcn_w  = (const float*)d_in[7];
    const float* gcn_b  = (const float*)d_in[8];
    const float* bn2_g  = (const float*)d_in[9];
    const float* bn2_b  = (const float*)d_in[10];
    const float* pool_w = (const float*)d_in[11];
    const int* src = ei;
    const int* dst = ei + N_EDGES;

    char*  base = (char*)d_ws;
    size_t off  = 0;
    auto alloc = [&](size_t bytes) { char* p = base + off; off = (off + bytes + 255) & ~(size_t)255; return p; };

    float* y     = (float*)alloc((size_t)N_NODES * C * 4);
    float* hw    = (float*)alloc((size_t)N_NODES * C * 4);
    float* agg   = (float*)alloc((size_t)N_NODES * C * 4);
    float* h2    = (float*)alloc((size_t)N_NODES * C * 4);
    float* deg   = (float*)alloc(N_NODES * 4);
    float* score = (float*)alloc(N_NODES * 4);
    float* stats = (float*)alloc(256 * 4);
    float* a1    = (float*)alloc(C * 4);
    float* s1    = (float*)alloc(C * 4);
    float* a2    = (float*)alloc(C * 4);
    float* s2    = (float*)alloc(C * 4);
    float* invnorm = (float*)alloc(4);
    int*   new_idx = (int*)alloc(N_NODES * 4);
    unsigned long long* keys = (unsigned long long*)alloc(SORT_N * 8);
    __bf16* xbf  = (__bf16*)alloc((size_t)N_NODES * C * 2);   // activations (reused)
    __bf16* wt   = (__bf16*)alloc(128 * WPITCH * 2);          // padded transposed weights

    float* out = (float*)d_out;
    const int GB = (N_NODES + TILE_ROWS - 1) / TILE_ROWS;

    // 1) y = x @ lin_w + lin_b                        (TDM-staged WMMA)
    conv_bf16<<<(N_NODES * C + 255) / 256, 256, 0, stream>>>(x, xbf, N_NODES * C);
    conv_wT<<<(128 * WPITCH + 255) / 256, 256, 0, stream>>>(lin_w, wt);
    gemm_tdm_wmma<<<GB, 128, 0, stream>>>(xbf, wt, lin_b, y, N_NODES);

    // 2) BN1 params
    fill_f32<<<1, 256, 0, stream>>>(stats, 0.0f, 256);
    colstats<<<256, 256, 0, stream>>>(y, nullptr, N_NODES, stats);
    bn_params<<<1, C, 0, stream>>>(stats, bn1_g, bn1_b, 1.0f / N_NODES, a1, s1);

    // 3) hw = relu(bn1(y)) @ gcn_w                    (BN+ReLU fused into conversion)
    conv_affine_bf16<<<(N_NODES * C + 255) / 256, 256, 0, stream>>>(y, a1, s1, xbf, N_NODES * C);
    conv_wT<<<(128 * WPITCH + 255) / 256, 256, 0, stream>>>(gcn_w, wt);
    gemm_tdm_wmma<<<GB, 128, 0, stream>>>(xbf, wt, nullptr, hw, N_NODES);

    // 4) degrees (self-loop counted as 1)
    fill_f32<<<(N_NODES + 255) / 256, 256, 0, stream>>>(deg, 1.0f, N_NODES);
    deg_accum<<<(N_EDGES + 255) / 256, 256, 0, stream>>>(dst, deg, N_EDGES);

    // 5) aggregation
    agg_init<<<(N_NODES * C + 255) / 256, 256, 0, stream>>>(hw, deg, agg, N_NODES * C);
    agg_edges<<<(N_EDGES * 32 + 255) / 256, 256, 0, stream>>>(src, dst, hw, deg, agg, N_EDGES);

    // 6) BN2 params over (agg + gcn_b); pool-vector norm
    fill_f32<<<1, 256, 0, stream>>>(stats, 0.0f, 256);
    colstats<<<256, 256, 0, stream>>>(agg, gcn_b, N_NODES, stats);
    bn_params<<<1, C, 0, stream>>>(stats, bn2_g, bn2_b, 1.0f / N_NODES, a2, s2);
    pool_norm_k<<<1, C, 0, stream>>>(pool_w, invnorm);

    // 7) h2 + score
    h2score<<<N_NODES, C, 0, stream>>>(agg, gcn_b, a2, s2, pool_w, invnorm, h2, score);

    // 8) sorted top-k (bitonic on 64-bit keys)
    build_keys<<<SORT_N / 256, 256, 0, stream>>>(score, keys);
    for (int k = 2; k <= SORT_N; k <<= 1)
        for (int j = k >> 1; j > 0; j >>= 1)
            bitonic_pass<<<SORT_N / 256, 256, 0, stream>>>(keys, j, k);

    // 9) selection + edge filtering
    fill_i32<<<(N_NODES + 255) / 256, 256, 0, stream>>>(new_idx, -1, N_NODES);
    select_topk<<<K_TOP, C, 0, stream>>>(keys, score, h2, batch, out, new_idx);
    edge_out<<<(N_EDGES + 255) / 256, 256, 0, stream>>>(src, dst, new_idx, out);
}
